// SALAD_24713241822082
// MI455X (gfx1250) — matrix operand order repaired and checked by reading, compiled, not run on hardware
//
#include <hip/hip_runtime.h>
#include <hip/hip_bf16.h>
#include <math.h>

typedef _Float16 v16h __attribute__((ext_vector_type(16)));
typedef float    v8f  __attribute__((ext_vector_type(8)));
typedef _Float16 h2v  __attribute__((ext_vector_type(2)));

#define DEV __device__ __forceinline__

static constexpr int BB = 16;    // batch
static constexpr int CC = 1536;  // channels
static constexpr int NN = 1024;  // pixels (32*32)
static constexpr int HH = 1024;  // stacked hidden (512 cluster + 512 score)
static constexpr int LL = 128;   // cluster dim
static constexpr int MM = 64;    // clusters
static constexpr int GG = 256;   // token dim
static constexpr int OUTROW = GG + LL * MM;  // 8448

DEV v8f zero8() {
  v8f z;
#pragma unroll
  for (int i = 0; i < 8; ++i) z[i] = 0.0f;
  return z;
}

DEV v8f wmma_f16(v16h a, v16h b, v8f c) {
  return __builtin_amdgcn_wmma_f32_16x16x32_f16(false, a, false, b, (short)0, c,
                                                false, false);
}

// A fragment: 16x32 f16, row-major source with leading dim ld.
// ISA layout: lane m = l&15; half (l>>4) picks K base 0/8; VGPR v: K = (v>>2)*16 + kb + 2*(v&3) + {0,1}
DEV v16h load_a_frag(const _Float16* A, int ld, int m0, int k0, int lane) {
  const _Float16* row = A + (size_t)(m0 + (lane & 15)) * ld + k0;
  int kb = (lane >> 4) * 8;
  v16h r;
#pragma unroll
  for (int v = 0; v < 8; ++v) {
    int kk = ((v >> 2) * 16) + kb + ((v & 3) * 2);
    h2v p = *(const h2v*)(row + kk);
    r[2 * v] = p[0];
    r[2 * v + 1] = p[1];
  }
  return r;
}

// B fragment: 32x16 f16 from B^T row-major storage Bt[col][k] (K contiguous per column).
// ISA layout: lane col = l&15; half (l>>4) picks K base 0/8; VGPR v: K = (v>>2)*16 + kb + 2*(v&3) + {0,1}
DEV v16h load_bt_frag(const _Float16* Bt, int ld, int n0, int k0, int lane) {
  const _Float16* row =
      Bt + (size_t)(n0 + (lane & 15)) * ld + k0 + ((lane >> 4) * 8);
  v16h r;
#pragma unroll
  for (int v = 0; v < 8; ++v) {
    h2v p = *(const h2v*)(row + ((v >> 2) * 16) + ((v & 3) * 2));
    r[2 * v] = p[0];
    r[2 * v + 1] = p[1];
  }
  return r;
}

// ---------------- pack kernels ----------------

__global__ void pack_x_kernel(const float* __restrict__ x, _Float16* __restrict__ xh) {
  int idx = blockIdx.x * 256 + threadIdx.x;  // < B*C*N = 25165824
  int n = idx & (NN - 1);
  int r = idx >> 10;  // b*C + c
  int c = r % CC;
  int b = r / CC;
  xh[((size_t)b * NN + n) * CC + c] = (_Float16)x[idx];
}

__global__ void pack_w1_kernel(const float* __restrict__ Wc1, const float* __restrict__ Ws1,
                               _Float16* __restrict__ W1h) {
  int idx = blockIdx.x * 256 + threadIdx.x;  // < 1024*1536
  int o = idx / CC;
  float v = (o < 512) ? Wc1[idx] : Ws1[idx - 512 * CC];
  W1h[idx] = (_Float16)v;
}

__global__ void pack_w2_kernel(const float* __restrict__ Wc2, const float* __restrict__ Ws2,
                               _Float16* __restrict__ W2h) {
  int idx = blockIdx.x * 256 + threadIdx.x;  // < 192*512
  int o = idx >> 9;
  float v = (o < 128) ? Wc2[idx] : Ws2[idx - 128 * 512];
  W2h[idx] = (_Float16)v;
}

__global__ void pack_bias_kernel(const float* bc1, const float* bs1, const float* bc2,
                                 const float* bs2, float* b1s, float* b2s) {
  int idx = blockIdx.x * 256 + threadIdx.x;
  if (idx < 1024) {
    b1s[idx] = (idx < 512) ? bc1[idx] : bs1[idx - 512];
  } else if (idx < 1024 + 192) {
    int j = idx - 1024;
    b2s[j] = (j < 128) ? bc2[j] : bs2[j - 128];
  }
}

// ---------------- token MLP (tiny, scalar) ----------------

__global__ void token_kernel(const float* __restrict__ t_in, const float* __restrict__ Wt1,
                             const float* __restrict__ bt1, const float* __restrict__ Wt2,
                             const float* __restrict__ bt2, float* __restrict__ tok) {
  int b = blockIdx.x, t = threadIdx.x;  // blockDim = 512
  __shared__ float ts[CC];
  __shared__ float h1[512];
  for (int i = t; i < CC; i += 512) ts[i] = t_in[(size_t)b * CC + i];
  __syncthreads();
  {
    const float* wr = Wt1 + (size_t)t * CC;
    float s = bt1[t];
    for (int c = 0; c < CC; ++c) s += ts[c] * wr[c];
    h1[t] = fmaxf(s, 0.0f);
  }
  __syncthreads();
  if (t < GG) {
    const float* wr = Wt2 + (size_t)t * 512;
    float s = bt2[t];
    for (int h = 0; h < 512; ++h) s += h1[h] * wr[h];
    tok[(size_t)b * GG + t] = s;
  }
}

// ---------------- GEMM1: h = relu(W1[1024x1536] * x[b]) -> h_t[b][n][o] f16 ----------------
// 32x32 register tile per wave: 2 A frags + 2 B frags feed 4 WMMAs per k-step.

DEV void store_h_tile(_Float16* __restrict__ h_t, const float* __restrict__ b1s,
                      int b, int n0, int o0, v8f acc, int col, int rb) {
  _Float16* orow = h_t + ((size_t)b * NN + (n0 + col)) * HH + o0 + rb;
#pragma unroll
  for (int v = 0; v < 8; v += 2) {
    float x0 = fmaxf(acc[v] + b1s[o0 + rb + v], 0.0f);
    float x1 = fmaxf(acc[v + 1] + b1s[o0 + rb + v + 1], 0.0f);
    h2v h;
    h[0] = (_Float16)x0;
    h[1] = (_Float16)x1;
    *(h2v*)(orow + v) = h;
  }
}

__global__ void gemm1_kernel(const _Float16* __restrict__ W1h, const _Float16* __restrict__ xh,
                             const float* __restrict__ b1s, _Float16* __restrict__ h_t) {
  int wave = threadIdx.x >> 5, lane = threadIdx.x & 31;  // blockDim = 128
  int o0 = blockIdx.x * 64 + (wave >> 1) * 32;
  int n0 = blockIdx.y * 64 + (wave & 1) * 32;
  int b = blockIdx.z;
  const _Float16* Bt = xh + (size_t)b * NN * CC;
  v8f acc[2][2];
#pragma unroll
  for (int i = 0; i < 2; ++i)
#pragma unroll
    for (int j = 0; j < 2; ++j) acc[i][j] = zero8();
  for (int k0 = 0; k0 < CC; k0 += 32) {
    v16h a0 = load_a_frag(W1h, CC, o0, k0, lane);
    v16h a1 = load_a_frag(W1h, CC, o0 + 16, k0, lane);
    v16h bf0 = load_bt_frag(Bt, CC, n0, k0, lane);
    v16h bf1 = load_bt_frag(Bt, CC, n0 + 16, k0, lane);
    acc[0][0] = wmma_f16(a0, bf0, acc[0][0]);
    acc[0][1] = wmma_f16(a0, bf1, acc[0][1]);
    acc[1][0] = wmma_f16(a1, bf0, acc[1][0]);
    acc[1][1] = wmma_f16(a1, bf1, acc[1][1]);
  }
  int col = lane & 15, rb = (lane >> 4) * 8;
#pragma unroll
  for (int i = 0; i < 2; ++i)
#pragma unroll
    for (int j = 0; j < 2; ++j)
      store_h_tile(h_t, b1s, b, n0 + j * 16, o0 + i * 16, acc[i][j], col, rb);
}

// ---------------- GEMM2: rows<128 -> f (two layouts), rows>=128 -> p (f32) ----------------
// 32x32 register tile per wave; o-tile of 32 keeps the f/p split uniform (128 % 32 == 0).

__global__ void gemm2_kernel(const _Float16* __restrict__ W2h, const _Float16* __restrict__ h_t,
                             const float* __restrict__ b2s, _Float16* __restrict__ fln,
                             _Float16* __restrict__ ftr, float* __restrict__ p32) {
  int lane = threadIdx.x & 31;  // blockDim = 32
  int o0 = blockIdx.x * 32;
  int n0 = blockIdx.y * 32;
  int b = blockIdx.z;
  int kbase = (o0 < 128) ? 0 : 512;  // f uses h1 (o 0..511), p uses h2 (o 512..1023)
  const _Float16* Bt = h_t + (size_t)b * NN * HH;
  v8f acc[2][2];
#pragma unroll
  for (int i = 0; i < 2; ++i)
#pragma unroll
    for (int j = 0; j < 2; ++j) acc[i][j] = zero8();
  for (int k0 = 0; k0 < 512; k0 += 32) {
    v16h a0 = load_a_frag(W2h, 512, o0, k0, lane);
    v16h a1 = load_a_frag(W2h, 512, o0 + 16, k0, lane);
    v16h bf0 = load_bt_frag(Bt, HH, n0, kbase + k0, lane);
    v16h bf1 = load_bt_frag(Bt, HH, n0 + 16, kbase + k0, lane);
    acc[0][0] = wmma_f16(a0, bf0, acc[0][0]);
    acc[0][1] = wmma_f16(a0, bf1, acc[0][1]);
    acc[1][0] = wmma_f16(a1, bf0, acc[1][0]);
    acc[1][1] = wmma_f16(a1, bf1, acc[1][1]);
  }
  int col = lane & 15, rb = (lane >> 4) * 8;
#pragma unroll
  for (int i = 0; i < 2; ++i) {
#pragma unroll
    for (int j = 0; j < 2; ++j) {
      int ob = o0 + i * 16, nb = n0 + j * 16;
      int n_abs = nb + col;
#pragma unroll
      for (int v = 0; v < 8; ++v) {
        int o_abs = ob + rb + v;
        float val = acc[i][j][v] + b2s[o_abs];
        if (o0 < 128) {
          fln[((size_t)b * LL + o_abs) * NN + n_abs] = (_Float16)val;
          ftr[((size_t)b * NN + n_abs) * LL + o_abs] = (_Float16)val;
        } else {
          p32[((size_t)b * MM + (o_abs - 128)) * NN + n_abs] = val;
        }
      }
    }
  }
}

// ---------------- Sinkhorn OT (per batch, in place p -> P) ----------------

__global__ void sinkhorn_kernel(float* __restrict__ p32, const float* __restrict__ dustp) {
  int b = blockIdx.x, t = threadIdx.x;  // blockDim = 256
  float* pb = p32 + (size_t)b * MM * NN;
  __shared__ float u_s[MM + 1];
  __shared__ float v_s[NN];
  float alpha = *dustp;
  const float norm = -logf((float)(MM + NN));
  for (int i = t; i < NN; i += 256) v_s[i] = 0.0f;
  __syncthreads();
  for (int it = 0; it < 3; ++it) {
    if (t < MM + 1) {
      float mx = -INFINITY;
      for (int n = 0; n < NN; ++n) {
        float z = ((t < MM) ? pb[(size_t)t * NN + n] : alpha) + v_s[n];
        mx = fmaxf(mx, z);
      }
      float s = 0.0f;
      for (int n = 0; n < NN; ++n) {
        float z = ((t < MM) ? pb[(size_t)t * NN + n] : alpha) + v_s[n];
        s += expf(z - mx);
      }
      float lmu = (t < MM) ? norm : (logf((float)NN) + norm);
      u_s[t] = lmu - (mx + logf(s));
    }
    __syncthreads();
    for (int q = 0; q < 4; ++q) {
      int n = t + q * 256;
      float mx = -INFINITY;
      for (int m = 0; m <= MM; ++m) {
        float z = ((m < MM) ? pb[(size_t)m * NN + n] : alpha) + u_s[m];
        mx = fmaxf(mx, z);
      }
      float s = 0.0f;
      for (int m = 0; m <= MM; ++m) {
        float z = ((m < MM) ? pb[(size_t)m * NN + n] : alpha) + u_s[m];
        s += expf(z - mx);
      }
      v_s[n] = norm - (mx + logf(s));
    }
    __syncthreads();
  }
  for (int i = t; i < MM * NN; i += 256) {
    int m = i >> 10, n = i & (NN - 1);
    pb[i] = expf(pb[i] + u_s[m] + v_s[n] - norm);
  }
}

// ---------------- T_i (softplus) + f_norm ----------------

__global__ void tfeat_kernel(const _Float16* __restrict__ ftr, const float* __restrict__ Wtp,
                             const float* __restrict__ btpp, float* __restrict__ Ti,
                             _Float16* __restrict__ fnt) {
  int idx = blockIdx.x * blockDim.x + threadIdx.x;  // b*N + n
  if (idx >= BB * NN) return;
  const _Float16* row = ftr + (size_t)idx * LL;
  float dot = 0.0f, ss = 0.0f;
  for (int l = 0; l < LL; ++l) {
    float v = (float)row[l];
    dot += v * Wtp[l];
    ss += v * v;
  }
  float xv = dot + *btpp;
  Ti[idx] = (xv > 20.0f) ? xv : log1pf(expf(xv));
  float rn = 1.0f / fmaxf(sqrtf(ss), 1e-12f);
  _Float16* o = fnt + (size_t)idx * LL;
  for (int l = 0; l < LL; ++l) o[l] = (_Float16)((float)row[l] * rn);
}

// ---------------- burst discount: WMMA similarity, fused sigmoid row-sum ----------------
// 32 i-rows per wave: A fragments stay resident across all 64 j-tiles; each B fragment
// feeds two WMMAs.

__global__ void discount_kernel(const _Float16* __restrict__ fnt, const float* __restrict__ Ti,
                                const float* __restrict__ bbp, const float* __restrict__ bpp,
                                float* __restrict__ dsc) {
  int i0 = blockIdx.x * 32, b = blockIdx.y;
  int lane = threadIdx.x & 31;  // blockDim = 32
  const _Float16* F = fnt + (size_t)b * NN * LL;
  const float* Tb = Ti + (size_t)b * NN;
  float bbv = *bbp, bpv = *bpp;
  v16h a[2][4];
#pragma unroll
  for (int i = 0; i < 2; ++i)
#pragma unroll
    for (int k = 0; k < 4; ++k) a[i][k] = load_a_frag(F, LL, i0 + i * 16, k * 32, lane);
  int col = lane & 15, rb = (lane >> 4) * 8;
  float Trow[2][8];
#pragma unroll
  for (int i = 0; i < 2; ++i)
#pragma unroll
    for (int v = 0; v < 8; ++v) Trow[i][v] = Tb[i0 + i * 16 + rb + v];
  float rs[2][8];
#pragma unroll
  for (int i = 0; i < 2; ++i)
#pragma unroll
    for (int v = 0; v < 8; ++v) rs[i][v] = 0.0f;
  for (int j0 = 0; j0 < NN; j0 += 16) {
    v8f acc0 = zero8(), acc1 = zero8();
#pragma unroll
    for (int k = 0; k < 4; ++k) {
      v16h bf = load_bt_frag(F, LL, j0, k * 32, lane);
      acc0 = wmma_f16(a[0][k], bf, acc0);
      acc1 = wmma_f16(a[1][k], bf, acc1);
    }
    float Tc = Tb[j0 + col];
#pragma unroll
    for (int v = 0; v < 8; ++v) {
      float x0 = 0.5f * (Trow[0][v] + Tc) * acc0[v] + bbv;
      rs[0][v] += 1.0f / (1.0f + expf(-x0));
      float x1 = 0.5f * (Trow[1][v] + Tc) * acc1[v] + bbv;
      rs[1][v] += 1.0f / (1.0f + expf(-x1));
    }
  }
#pragma unroll
  for (int i = 0; i < 2; ++i) {
#pragma unroll
    for (int v = 0; v < 8; ++v) {  // sum across the 16 lanes of each half
      float s = rs[i][v];
      s += __shfl_xor(s, 1, 32);
      s += __shfl_xor(s, 2, 32);
      s += __shfl_xor(s, 4, 32);
      s += __shfl_xor(s, 8, 32);
      rs[i][v] = s;
    }
  }
  if (col == 0) {
#pragma unroll
    for (int i = 0; i < 2; ++i) {
#pragma unroll
      for (int v = 0; v < 8; ++v) {
        float d = powf(rs[i][v], bpv);
        d = fminf(fmaxf(d, 1e-3f), 1e3f);
        dsc[(size_t)b * NN + i0 + i * 16 + rb + v] = d;
      }
    }
  }
}

// ---------------- P_adj = P * ((1-lam) + lam/(disc+eps)) -> f16 ----------------

__global__ void padj_kernel(const float* __restrict__ P, const float* __restrict__ dsc,
                            const float* __restrict__ lamp, _Float16* __restrict__ Ph) {
  int idx = blockIdx.x * 256 + threadIdx.x;  // < B*M*N = 2^20
  float lam = 1.0f / (1.0f + expf(-*lamp));
  int n = idx & (NN - 1);
  int b = idx >> 16;  // M*N = 65536
  float w = (1.0f - lam) + lam / (dsc[(size_t)b * NN + n] + 1e-6f);
  Ph[idx] = (_Float16)(P[idx] * w);
}

// ---------------- agg = f[LxN] * P_adj^T[NxM] (WMMA) ----------------

__global__ void agg_kernel(const _Float16* __restrict__ fln, const _Float16* __restrict__ Ph,
                           float* __restrict__ agg) {
  int l0 = blockIdx.x * 16, m0 = blockIdx.y * 16, b = blockIdx.z;
  int lane = threadIdx.x & 31;  // blockDim = 32
  const _Float16* A = fln + (size_t)b * LL * NN;
  const _Float16* Bt = Ph + (size_t)b * MM * NN;
  v8f acc = zero8();
  for (int k0 = 0; k0 < NN; k0 += 32) {
    v16h a = load_a_frag(A, NN, l0, k0, lane);
    v16h bf = load_bt_frag(Bt, NN, m0, k0, lane);
    acc = wmma_f16(a, bf, acc);
  }
  int col = lane & 15, rb = (lane >> 4) * 8;
#pragma unroll
  for (int v = 0; v < 8; ++v) {
    agg[((size_t)b * LL + (l0 + rb + v)) * MM + (m0 + col)] = acc[v];
  }
}

// ---------------- finalize: column norms, tok norm, global l2 ----------------

__global__ void finalize_kernel(const float* __restrict__ tok, const float* __restrict__ agg,
                                float* __restrict__ out) {
  int b = blockIdx.x, t = threadIdx.x;  // blockDim = 256
  __shared__ float red[256];
  __shared__ float cn[MM];
  __shared__ float sh_tinv, sh_ginv;
  const float* tb = tok + (size_t)b * GG;
  const float* ab = agg + (size_t)b * LL * MM;
  float tv = tb[t];
  red[t] = tv * tv;
  __syncthreads();
  for (int s = 128; s > 0; s >>= 1) {
    if (t < s) red[t] += red[t + s];
    __syncthreads();
  }
  if (t == 0) sh_tinv = 1.0f / fmaxf(sqrtf(red[0]), 1e-12f);
  if (t < MM) {
    float s = 0.0f;
    for (int l = 0; l < LL; ++l) {
      float v = ab[l * MM + t];
      s += v * v;
    }
    cn[t] = fmaxf(sqrtf(s), 1e-12f);
  }
  __syncthreads();
  float tn = tv * sh_tinv;
  float part = tn * tn;
  for (int i = t; i < LL * MM; i += 256) {
    int m = i & (MM - 1);
    float v = ab[i] / cn[m];
    part += v * v;
  }
  red[t] = part;
  __syncthreads();
  for (int s = 128; s > 0; s >>= 1) {
    if (t < s) red[t] += red[t + s];
    __syncthreads();
  }
  if (t == 0) sh_ginv = 1.0f / fmaxf(sqrtf(red[0]), 1e-12f);
  __syncthreads();
  float ginv = sh_ginv;
  float* ob = out + (size_t)b * OUTROW;
  ob[t] = tn * ginv;
  for (int i = t; i < LL * MM; i += 256) {
    int m = i & (MM - 1);
    ob[GG + i] = (ab[i] / cn[m]) * ginv;
  }
}

// ---------------- launch ----------------

extern "C" void kernel_launch(void* const* d_in, const int* in_sizes, int n_in,
                              void* d_out, int out_size, void* d_ws, size_t ws_size,
                              hipStream_t stream) {
  const float* x = (const float*)d_in[0];
  const float* t = (const float*)d_in[1];
  const float* Wt1 = (const float*)d_in[2];
  const float* bt1 = (const float*)d_in[3];
  const float* Wt2 = (const float*)d_in[4];
  const float* bt2 = (const float*)d_in[5];
  const float* Wc1 = (const float*)d_in[6];
  const float* bc1 = (const float*)d_in[7];
  const float* Wc2 = (const float*)d_in[8];
  const float* bc2 = (const float*)d_in[9];
  const float* Ws1 = (const float*)d_in[10];
  const float* bs1 = (const float*)d_in[11];
  const float* Ws2 = (const float*)d_in[12];
  const float* bs2 = (const float*)d_in[13];
  const float* Wtp = (const float*)d_in[14];
  const float* btp = (const float*)d_in[15];
  const float* dust = (const float*)d_in[16];
  const float* bb = (const float*)d_in[17];
  const float* bp = (const float*)d_in[18];
  const float* lam = (const float*)d_in[19];
  (void)in_sizes; (void)n_in; (void)out_size; (void)ws_size;

  char* ws = (char*)d_ws;
  size_t off = 0;
  auto alloc = [&](size_t bytes) -> char* {
    char* p = ws + off;
    off += (bytes + 255) & ~(size_t)255;
    return p;
  };
  _Float16* xh = (_Float16*)alloc((size_t)BB * NN * CC * 2);
  _Float16* W1h = (_Float16*)alloc((size_t)HH * CC * 2);
  _Float16* W2h = (_Float16*)alloc((size_t)192 * 512 * 2);
  float* b1s = (float*)alloc(1024 * 4);
  float* b2s = (float*)alloc(192 * 4);
  _Float16* h_t = (_Float16*)alloc((size_t)BB * NN * HH * 2);
  _Float16* fln = (_Float16*)alloc((size_t)BB * LL * NN * 2);
  _Float16* ftr = (_Float16*)alloc((size_t)BB * NN * LL * 2);
  _Float16* fnt = (_Float16*)alloc((size_t)BB * NN * LL * 2);
  float* p32 = (float*)alloc((size_t)BB * MM * NN * 4);
  _Float16* Ph = (_Float16*)alloc((size_t)BB * MM * NN * 2);
  float* Ti = (float*)alloc((size_t)BB * NN * 4);
  float* dsc = (float*)alloc((size_t)BB * NN * 4);
  float* tokb = (float*)alloc((size_t)BB * GG * 4);
  float* aggb = (float*)alloc((size_t)BB * LL * MM * 4);

  pack_x_kernel<<<(BB * CC * NN) / 256, 256, 0, stream>>>(x, xh);
  pack_w1_kernel<<<(HH * CC) / 256, 256, 0, stream>>>(Wc1, Ws1, W1h);
  pack_w2_kernel<<<(192 * 512) / 256, 256, 0, stream>>>(Wc2, Ws2, W2h);
  pack_bias_kernel<<<5, 256, 0, stream>>>(bc1, bs1, bc2, bs2, b1s, b2s);
  token_kernel<<<BB, 512, 0, stream>>>(t, Wt1, bt1, Wt2, bt2, tokb);
  gemm1_kernel<<<dim3(HH / 64, NN / 64, BB), 128, 0, stream>>>(W1h, xh, b1s, h_t);
  gemm2_kernel<<<dim3(192 / 32, NN / 32, BB), 32, 0, stream>>>(W2h, h_t, b2s, fln, ftr, p32);
  sinkhorn_kernel<<<BB, 256, 0, stream>>>(p32, dust);
  tfeat_kernel<<<(BB * NN) / 128, 128, 0, stream>>>(ftr, Wtp, btp, Ti, fnt);
  discount_kernel<<<dim3(NN / 32, BB), 32, 0, stream>>>(fnt, Ti, bb, bp, dsc);
  padj_kernel<<<(BB * MM * NN) / 256, 256, 0, stream>>>(p32, dsc, lam, Ph);
  agg_kernel<<<dim3(LL / 16, MM / 16, BB), 32, 0, stream>>>(fln, Ph, aggb);
  finalize_kernel<<<BB, 256, 0, stream>>>(tokb, aggb, (float*)d_out);
}
